// EngineA_51891794870678
// MI455X (gfx1250) — matrix operand order repaired and checked by reading, not compiled, not run on hardware
//
#include <hip/hip_runtime.h>
#include <hip/hip_bf16.h>

#define N_TOKENS 8192
#define DIM_IN   1024
#define DIM_HID  4096
#define DIM_OUT  1024
#define N_EXP    12
#define TOPK     4

typedef __attribute__((ext_vector_type(16))) __bf16       bf16x16;
typedef __attribute__((ext_vector_type(8)))  float        f32x8;
typedef __attribute__((ext_vector_type(4)))  float        f32x4;
typedef __attribute__((ext_vector_type(4)))  unsigned int u32x4;

union Frag  { u32x4 q[2]; bf16x16 v; };
union Pack8 { u32x4 q; unsigned short s[8]; };

__device__ __forceinline__ unsigned short f2bf(float f) {
  unsigned int u = __float_as_uint(f);
  u += 0x7FFFu + ((u >> 16) & 1u);   // round-to-nearest-even
  return (unsigned short)(u >> 16);
}

// ---------------- init: zero expert counts ----------------
__global__ void moe_init(int* counts) {
  if (threadIdx.x < 16) counts[threadIdx.x] = 0;
}

// ---------------- f32 -> bf16 conversion ----------------
__global__ void moe_cvt_bf16(const float* __restrict__ src,
                             unsigned short* __restrict__ dst, long n4) {
  long i = (long)blockIdx.x * blockDim.x + threadIdx.x;
  long stride = (long)gridDim.x * blockDim.x;
  for (; i < n4; i += stride) {
    f32x4 v = ((const f32x4*)src)[i];
    unsigned int lo = (unsigned)f2bf(v.x) | ((unsigned)f2bf(v.y) << 16);
    unsigned int hi = (unsigned)f2bf(v.z) | ((unsigned)f2bf(v.w) << 16);
    ((unsigned int*)dst)[2 * i]     = lo;
    ((unsigned int*)dst)[2 * i + 1] = hi;
  }
}

// ---------------- gate: top-4 softmax routing, 1 wave32 per token ----------------
__global__ void moe_route(const float* __restrict__ x, const float* __restrict__ Wg,
                          const float* __restrict__ bg, int* __restrict__ counts,
                          int* __restrict__ tok_by_e, int* __restrict__ assign,
                          float* __restrict__ wts) {
  int token = blockIdx.x * 4 + (threadIdx.x >> 5);
  int lane  = threadIdx.x & 31;
  const float* xr = x + (size_t)token * DIM_IN;
  float acc[N_EXP];
#pragma unroll
  for (int e = 0; e < N_EXP; ++e) acc[e] = 0.f;
  for (int k = lane; k < DIM_IN; k += 32) {
    float xv = xr[k];
    const float* wr = Wg + (size_t)k * N_EXP;
#pragma unroll
    for (int e = 0; e < N_EXP; ++e) acc[e] += xv * wr[e];
  }
#pragma unroll
  for (int off = 16; off > 0; off >>= 1) {
#pragma unroll
    for (int e = 0; e < N_EXP; ++e) acc[e] += __shfl_xor(acc[e], off, 32);
  }
  if (lane == 0) {
    float lg[N_EXP];
#pragma unroll
    for (int e = 0; e < N_EXP; ++e) lg[e] = acc[e] + bg[e];
    int ti[TOPK]; float tv[TOPK];
    unsigned used = 0;
    for (int t = 0; t < TOPK; ++t) {
      float best = -3.4e38f; int bi = 0;
      for (int e = 0; e < N_EXP; ++e)
        if (!((used >> e) & 1u) && lg[e] > best) { best = lg[e]; bi = e; }
      used |= 1u << bi; ti[t] = bi; tv[t] = best;
    }
    float ex[TOPK], s = 0.f;
    for (int t = 0; t < TOPK; ++t) { ex[t] = __expf(tv[t] - tv[0]); s += ex[t]; }
    float inv = 1.f / s;
    for (int t = 0; t < TOPK; ++t) {
      int e = ti[t];
      int slot = atomicAdd(&counts[e], 1);
      tok_by_e[e * N_TOKENS + slot] = token;
      assign[token * TOPK + t] = e * N_TOKENS + slot;  // N_TOKENS = 2^13
      wts[token * TOPK + t]    = ex[t] * inv;
    }
  }
}

// ---------------- exclusive prefix of counts ----------------
__global__ void moe_offsets(const int* __restrict__ counts, int* __restrict__ offs) {
  if (threadIdx.x == 0) {
    int s = 0;
    for (int e = 0; e < N_EXP; ++e) { offs[e] = s; s += counts[e]; }
  }
}

// ---------------- GEMM1: H = relu(gather(X) @ W1[e] + b1[e]) ----------------
// 128x128 tile, BK=32, 256 threads = 8 waves; each wave: 2(M) x 4(N) wmma tiles.
// Software-pipelined: next tile is staged in VGPRs while WMMAs run on current LDS tile.
__global__ __launch_bounds__(256) void moe_gemm1(
    const unsigned short* __restrict__ X16, const unsigned short* __restrict__ W116,
    const float* __restrict__ b1, const int* __restrict__ tok_by_e,
    const int* __restrict__ counts, const int* __restrict__ offs,
    unsigned short* __restrict__ H) {
  const int e   = blockIdx.z;
  const int cnt = counts[e];
  const int m0  = blockIdx.x * 128;
  if (m0 >= cnt) return;
  const int n0 = blockIdx.y * 128;
  const unsigned short* W = W116 + (size_t)e * DIM_IN * DIM_HID;
  const int* toks = tok_by_e + e * N_TOKENS;
  const int slot0 = offs[e];

  __shared__ __align__(16) unsigned short Xs[128 * 32];
  __shared__ __align__(16) unsigned short Ws[128 * 32];

  const int tid = threadIdx.x;
  // ---- k-invariant loader coordinates (each thread owns 2 chunks per tile) ----
  const int xr0 = tid >> 2;            // rows xr0 and xr0+64
  const int xo0 = (tid & 3) * 8;       // element offset within 32-wide k-slice
  const int kk0 = tid >> 4;            // W rows kk0 and kk0+16
  const int nc0 = (tid & 15) * 8;      // 8 consecutive n per chunk
  // clamped gather rows: invalid rows read a valid address; their C-rows are discarded
  const int gr0 = (m0 + xr0 < cnt) ? (m0 + xr0) : (cnt - 1);
  const int gr1 = (m0 + xr0 + 64 < cnt) ? (m0 + xr0 + 64) : (cnt - 1);
  const unsigned short* xp0 = X16 + (size_t)toks[gr0] * DIM_IN + xo0;
  const unsigned short* xp1 = X16 + (size_t)toks[gr1] * DIM_IN + xo0;
  const unsigned short* wp0 = W + (size_t)kk0 * DIM_HID + n0 + nc0;
  const unsigned short* wp1 = W + (size_t)(kk0 + 16) * DIM_HID + n0 + nc0;

  const int wid = tid >> 5, lane = tid & 31;
  const int wm = (wid >> 1) * 32, wn = (wid & 1) * 64;
  const int half = lane >> 4, l15 = lane & 15;

  f32x8 acc[2][4];
#pragma unroll
  for (int i = 0; i < 2; ++i)
#pragma unroll
    for (int j = 0; j < 4; ++j)
#pragma unroll
      for (int t = 0; t < 8; ++t) acc[i][j][t] = 0.f;

  // prologue: stage tile k0=0
  u32x4 xa = *(const u32x4*)xp0;
  u32x4 xb = *(const u32x4*)xp1;
  Pack8 wa, wb;
  wa.q = *(const u32x4*)wp0;
  wb.q = *(const u32x4*)wp1;

  for (int k0 = 0; k0 < DIM_IN; k0 += 32) {
    // commit staged tile to LDS
    *(u32x4*)&Xs[xr0 * 32 + xo0]        = xa;
    *(u32x4*)&Xs[(xr0 + 64) * 32 + xo0] = xb;
#pragma unroll
    for (int j = 0; j < 8; ++j) Ws[(nc0 + j) * 32 + kk0]      = wa.s[j];
#pragma unroll
    for (int j = 0; j < 8; ++j) Ws[(nc0 + j) * 32 + kk0 + 16] = wb.s[j];
    __syncthreads();

    // stage next tile (stays in flight across the WMMA block)
    if (k0 + 32 < DIM_IN) {
      xa = *(const u32x4*)(xp0 + k0 + 32);
      xb = *(const u32x4*)(xp1 + k0 + 32);
      wa.q = *(const u32x4*)(wp0 + (size_t)(k0 + 32) * DIM_HID);
      wb.q = *(const u32x4*)(wp1 + (size_t)(k0 + 32) * DIM_HID);
    }

    Frag a[2], b[4];
#pragma unroll
    for (int i = 0; i < 2; ++i) {
      int base = (wm + i * 16 + l15) * 32 + half * 8;  // A: lane=M, K 0-7/16-23 vs 8-15/24-31
      a[i].q[0] = *(const u32x4*)&Xs[base];
      a[i].q[1] = *(const u32x4*)&Xs[base + 16];
    }
#pragma unroll
    for (int j = 0; j < 4; ++j) {
      int base = (wn + j * 16 + l15) * 32 + half * 8;  // B: lane=N, K 0-7/16-23 vs 8-15/24-31
      b[j].q[0] = *(const u32x4*)&Ws[base];
      b[j].q[1] = *(const u32x4*)&Ws[base + 16];
    }
#pragma unroll
    for (int i = 0; i < 2; ++i)
#pragma unroll
      for (int j = 0; j < 4; ++j)
        acc[i][j] = __builtin_amdgcn_wmma_f32_16x16x32_bf16(
            false, a[i].v, false, b[j].v, (short)0, acc[i][j], false, false);
    __syncthreads();
  }

  const float* b1e = b1 + (size_t)e * DIM_HID;
#pragma unroll
  for (int j = 0; j < 4; ++j) {
    int gn = n0 + wn + j * 16 + l15;
    float bias = b1e[gn];
#pragma unroll
    for (int i = 0; i < 2; ++i)
#pragma unroll
      for (int v = 0; v < 8; ++v) {
        int m = m0 + wm + i * 16 + half * 8 + v;       // C: lanes16-31 hold M=v+8
        if (m < cnt) {
          float h = acc[i][j][v] + bias;
          h = h > 0.f ? h : 0.f;
          H[(size_t)(slot0 + m) * DIM_HID + gn] = f2bf(h);
        }
      }
  }
}

// ---------------- GEMM2: O = H @ W2[e] + b2[e] ----------------
__global__ __launch_bounds__(256) void moe_gemm2(
    const unsigned short* __restrict__ H, const unsigned short* __restrict__ W216,
    const float* __restrict__ b2, const int* __restrict__ counts,
    const int* __restrict__ offs, float* __restrict__ Obuf) {
  const int e   = blockIdx.z;
  const int cnt = counts[e];
  const int m0  = blockIdx.x * 128;
  if (m0 >= cnt) return;
  const int n0 = blockIdx.y * 128;
  const unsigned short* W = W216 + (size_t)e * DIM_HID * DIM_OUT;
  const int slot0 = offs[e];

  __shared__ __align__(16) unsigned short As[128 * 32];
  __shared__ __align__(16) unsigned short Bs[128 * 32];

  const int tid = threadIdx.x;
  const int xr0 = tid >> 2;
  const int xo0 = (tid & 3) * 8;
  const int kk0 = tid >> 4;
  const int nc0 = (tid & 15) * 8;
  // rows beyond cnt stay inside the padded H buffer; their C-rows are discarded
  const unsigned short* ap0 = H + (size_t)(slot0 + m0 + xr0) * DIM_HID + xo0;
  const unsigned short* ap1 = H + (size_t)(slot0 + m0 + xr0 + 64) * DIM_HID + xo0;
  const unsigned short* wp0 = W + (size_t)kk0 * DIM_OUT + n0 + nc0;
  const unsigned short* wp1 = W + (size_t)(kk0 + 16) * DIM_OUT + n0 + nc0;

  const int wid = tid >> 5, lane = tid & 31;
  const int wm = (wid >> 1) * 32, wn = (wid & 1) * 64;
  const int half = lane >> 4, l15 = lane & 15;

  f32x8 acc[2][4];
#pragma unroll
  for (int i = 0; i < 2; ++i)
#pragma unroll
    for (int j = 0; j < 4; ++j)
#pragma unroll
      for (int t = 0; t < 8; ++t) acc[i][j][t] = 0.f;

  u32x4 xa = *(const u32x4*)ap0;
  u32x4 xb = *(const u32x4*)ap1;
  Pack8 wa, wb;
  wa.q = *(const u32x4*)wp0;
  wb.q = *(const u32x4*)wp1;

  for (int k0 = 0; k0 < DIM_HID; k0 += 32) {
    *(u32x4*)&As[xr0 * 32 + xo0]        = xa;
    *(u32x4*)&As[(xr0 + 64) * 32 + xo0] = xb;
#pragma unroll
    for (int j = 0; j < 8; ++j) Bs[(nc0 + j) * 32 + kk0]      = wa.s[j];
#pragma unroll
    for (int j = 0; j < 8; ++j) Bs[(nc0 + j) * 32 + kk0 + 16] = wb.s[j];
    __syncthreads();

    if (k0 + 32 < DIM_HID) {
      xa = *(const u32x4*)(ap0 + k0 + 32);
      xb = *(const u32x4*)(ap1 + k0 + 32);
      wa.q = *(const u32x4*)(wp0 + (size_t)(k0 + 32) * DIM_OUT);
      wb.q = *(const u32x4*)(wp1 + (size_t)(k0 + 32) * DIM_OUT);
    }

    Frag a[2], b[4];
#pragma unroll
    for (int i = 0; i < 2; ++i) {
      int base = (wm + i * 16 + l15) * 32 + half * 8;
      a[i].q[0] = *(const u32x4*)&As[base];
      a[i].q[1] = *(const u32x4*)&As[base + 16];
    }
#pragma unroll
    for (int j = 0; j < 4; ++j) {
      int base = (wn + j * 16 + l15) * 32 + half * 8;
      b[j].q[0] = *(const u32x4*)&Bs[base];
      b[j].q[1] = *(const u32x4*)&Bs[base + 16];
    }
#pragma unroll
    for (int i = 0; i < 2; ++i)
#pragma unroll
      for (int j = 0; j < 4; ++j)
        acc[i][j] = __builtin_amdgcn_wmma_f32_16x16x32_bf16(
            false, a[i].v, false, b[j].v, (short)0, acc[i][j], false, false);
    __syncthreads();
  }

  const float* b2e = b2 + (size_t)e * DIM_OUT;
#pragma unroll
  for (int j = 0; j < 4; ++j) {
    int gn = n0 + wn + j * 16 + l15;
    float bias = b2e[gn];
#pragma unroll
    for (int i = 0; i < 2; ++i)
#pragma unroll
      for (int v = 0; v < 8; ++v) {
        int m = m0 + wm + i * 16 + half * 8 + v;
        if (m < cnt)
          Obuf[(size_t)(slot0 + m) * DIM_OUT + gn] = acc[i][j][v] + bias;
      }
  }
}

// ---------------- combine: out[n] = sum_t w_t * O[row_t] (deterministic) ----------------
__global__ void moe_combine(const float* __restrict__ Obuf, const int* __restrict__ assign,
                            const float* __restrict__ wts, const int* __restrict__ offs,
                            float* __restrict__ out) {
  int gid = blockIdx.x * blockDim.x + threadIdx.x;  // N_TOKENS * 256
  int token = gid >> 8;
  int dc = gid & 255;
  f32x4 s; s.x = 0.f; s.y = 0.f; s.z = 0.f; s.w = 0.f;
#pragma unroll
  for (int t = 0; t < TOPK; ++t) {
    int av = assign[token * TOPK + t];
    int ee = av >> 13, slot = av & (N_TOKENS - 1);
    size_t row = (size_t)offs[ee] + slot;
    float w = wts[token * TOPK + t];
    f32x4 o = ((const f32x4*)(Obuf + row * DIM_OUT))[dc];
    s.x += w * o.x; s.y += w * o.y; s.z += w * o.z; s.w += w * o.w;
  }
  ((f32x4*)(out + (size_t)token * DIM_OUT))[dc] = s;
}

extern "C" void kernel_launch(void* const* d_in, const int* in_sizes, int n_in,
                              void* d_out, int out_size, void* d_ws, size_t ws_size,
                              hipStream_t stream) {
  const float* x  = (const float*)d_in[0];
  const float* W1 = (const float*)d_in[1];
  const float* b1 = (const float*)d_in[2];
  const float* W2 = (const float*)d_in[3];
  const float* b2 = (const float*)d_in[4];
  const float* Wg = (const float*)d_in[5];
  const float* bg = (const float*)d_in[6];
  float* out = (float*)d_out;
  (void)in_sizes; (void)n_in; (void)out_size; (void)ws_size;

  char* ws = (char*)d_ws;
  size_t off = 0;
  auto alloc = [&](size_t bytes) -> void* {
    void* p = ws + off;
    off = (off + bytes + 255) & ~(size_t)255;
    return p;
  };
  unsigned short* X16  = (unsigned short*)alloc((size_t)N_TOKENS * DIM_IN * 2);
  unsigned short* W116 = (unsigned short*)alloc((size_t)N_EXP * DIM_IN * DIM_HID * 2);
  unsigned short* W216 = (unsigned short*)alloc((size_t)N_EXP * DIM_HID * DIM_OUT * 2);
  int*   tok_by_e = (int*)alloc((size_t)N_EXP * N_TOKENS * 4);
  int*   assign   = (int*)alloc((size_t)N_TOKENS * TOPK * 4);
  float* wts      = (float*)alloc((size_t)N_TOKENS * TOPK * 4);
  int*   counts   = (int*)alloc(64);
  int*   offs     = (int*)alloc(64);
  unsigned short* H = (unsigned short*)alloc(((size_t)TOPK * N_TOKENS + 192) * DIM_HID * 2);
  float* Obuf       = (float*)alloc(((size_t)TOPK * N_TOKENS + 192) * DIM_OUT * 4);

  moe_init<<<1, 32, 0, stream>>>(counts);
  moe_cvt_bf16<<<2048, 256, 0, stream>>>(x,  X16,  (long)N_TOKENS * DIM_IN / 4);
  moe_cvt_bf16<<<4096, 256, 0, stream>>>(W1, W116, (long)N_EXP * DIM_IN * DIM_HID / 4);
  moe_cvt_bf16<<<4096, 256, 0, stream>>>(W2, W216, (long)N_EXP * DIM_HID * DIM_OUT / 4);
  moe_route<<<N_TOKENS / 4, 128, 0, stream>>>(x, Wg, bg, counts, tok_by_e, assign, wts);
  moe_offsets<<<1, 32, 0, stream>>>(counts, offs);
  moe_gemm1<<<dim3(64, 32, 12), 256, 0, stream>>>(X16, W116, b1, tok_by_e, counts, offs, H);
  moe_gemm2<<<dim3(64, 8, 12), 256, 0, stream>>>(H, W216, b2, counts, offs, Obuf);
  moe_combine<<<N_TOKENS, 256, 0, stream>>>(Obuf, assign, wts, offs, out);
}
